// PushNP_3143916061322
// MI455X (gfx1250) — matrix operand order repaired and checked by reading, not compiled, not run on hardware
//
#include <hip/hip_runtime.h>
#include <math.h>

// ---------------------------------------------------------------------------
// CDNA5 / gfx1250 PushNP forward pass, round 2.
// All GEMMs: v_wmma_f32_16x16x32_f16, f16 operands / f32 accumulate.
// Key change vs round 1: weights are pre-packed into the exact B-fragment
// layout ([n-tile][k-chunk][lane][16 halves], f16, zero padded) and all
// activations are stored as f16 with K padded to a multiple of 32, so the
// GEMM inner loop is completely branch/guard free:
//   per 32-wide k-chunk: 2x b128 (A) + 8x b128 (B, 4 tiles) + 4x WMMA.
// Max-pools remain fused into the GEMM epilogue via encoded u32 atomicMax.
// ---------------------------------------------------------------------------

typedef __attribute__((ext_vector_type(16))) _Float16 v16h;
typedef __attribute__((ext_vector_type(8)))  _Float16 v8h;
typedef __attribute__((ext_vector_type(8)))  float    v8f;

// ---- helpers ---------------------------------------------------------------
__device__ __forceinline__ unsigned enc_ord(float f) {
  unsigned u = __float_as_uint(f);
  return (u & 0x80000000u) ? ~u : (u | 0x80000000u);
}
__device__ __forceinline__ float dec_ord(unsigned e) {
  return __uint_as_float((e & 0x80000000u) ? (e ^ 0x80000000u) : ~e);
}
__device__ __forceinline__ float softplus_f(float x) {
  return fmaxf(x, 0.f) + log1pf(expf(-fabsf(x)));
}

// ---- weight pre-pack: W[N][K] f32 -> P[(tile*KC+kc)*32+lane][16] f16 -------
// B-fragment layout: lane%16 = N index within 16-wide tile, element h = K index
// (h<8: 8*(lane/16)+h, else 16+8*(lane/16)+h-8). Zero padded to KC*32 x NT*16.
__global__ void pack_w_kernel(const float* __restrict__ W, _Float16* __restrict__ P,
                              int K, int N, int KC, int NT) {
  int i = blockIdx.x * blockDim.x + threadIdx.x;
  if (i >= NT * KC * 32) return;
  int lane = i & 31;
  int rest = i >> 5;
  int kc   = rest % KC;
  int tile = rest / KC;
  int n = (tile << 4) + (lane & 15);
  int hh = lane >> 4;
  _Float16* o = P + (size_t)i * 16;
#pragma unroll
  for (int h = 0; h < 16; ++h) {
    int k = (kc << 5) + ((h < 8) ? (8 * hh + h) : (16 + 8 * hh + (h - 8)));
    float v = (n < N && k < K) ? W[(size_t)n * K + k] : 0.f;
    o[h] = (_Float16)v;
  }
}

// ---- input pad/convert: X[rows][K] f32 -> Y[rows][Kp] f16 (zero pad) -------
__global__ void pad_input_kernel(const float* __restrict__ X, _Float16* __restrict__ Y,
                                 int rows, int K, int Kp) {
  int r = blockIdx.x * blockDim.x + threadIdx.x;
  if (r >= rows) return;
  const float* x = X + (size_t)r * K;
  _Float16* y = Y + (size_t)r * Kp;
  for (int k = 0; k < Kp; ++k) y[k] = (k < K) ? (_Float16)x[k] : (_Float16)0.f;
}

// ---- WMMA GEMM:  act(A[RxK] * W^T + bias) ---------------------------------
// A: f16 row-major, K % 32 == 0 (padded).  Bpk: packed fragments (see above).
// store mode  (Cmax==null): Cout[row*ldc+col] as f16 (half_out) or f32
// reduce mode (Cmax!=null): encoded atomicMax over groups of (1<<gshift) rows
__global__ __launch_bounds__(256) void wmma_gemm_kernel(
    const _Float16* __restrict__ A, const _Float16* __restrict__ Bpk,
    const float* __restrict__ bias,
    void* __restrict__ Cout, unsigned* __restrict__ Cmax,
    int R, int K, int N, int ldc, int relu, int gshift, int half_out)
{
  const int CT = (N + 63) >> 6;            // 64-wide column supertiles
  const int RT = R >> 4;                   // R always a multiple of 16
  const int w  = blockIdx.x * 8 + (threadIdx.x >> 5);
  if (w >= RT * CT) return;                // wave-uniform
  const int rt    = w / CT;
  const int ct    = w - rt * CT;
  const int Mbase = rt << 4;
  const int Nsup  = ct << 6;
  const int lane  = threadIdx.x & 31;
  const int l16   = lane & 15;
  const int kb    = (lane & 16) ? 8 : 0;   // A: upper half-wave covers K+8 block
  const int rOff  = (lane & 16) ? 8 : 0;
  const int KC    = K >> 5;

  const _Float16* arow  = A + (size_t)(Mbase + l16) * K + kb;
  const _Float16* bbase = Bpk + ((size_t)(ct << 2) * KC * 32 + lane) * 16;
  const size_t    btile = (size_t)KC * 512;   // halves per 16-wide n-tile

  v8f acc0 = {}, acc1 = {}, acc2 = {}, acc3 = {};

  for (int kc = 0; kc < KC; ++kc) {
    const _Float16* ap = arow + (kc << 5);
    v8h lo = *(const v8h*)(ap);
    v8h hi = *(const v8h*)(ap + 16);
    v16h af = __builtin_shufflevector(lo, hi, 0,1,2,3,4,5,6,7,8,9,10,11,12,13,14,15);

    const _Float16* bp = bbase + (size_t)kc * 512;
    v16h b0 = *(const v16h*)(bp);
    v16h b1 = *(const v16h*)(bp + btile);
    v16h b2 = *(const v16h*)(bp + 2 * btile);
    v16h b3 = *(const v16h*)(bp + 3 * btile);

    acc0 = __builtin_amdgcn_wmma_f32_16x16x32_f16(false, af, false, b0, (short)0, acc0, false, false);
    acc1 = __builtin_amdgcn_wmma_f32_16x16x32_f16(false, af, false, b1, (short)0, acc1, false, false);
    acc2 = __builtin_amdgcn_wmma_f32_16x16x32_f16(false, af, false, b2, (short)0, acc2, false, false);
    acc3 = __builtin_amdgcn_wmma_f32_16x16x32_f16(false, af, false, b3, (short)0, acc3, false, false);
  }

  // C/D layout: VGPR r -> row Mbase + r (+8 for lanes 16-31), col = n0 + (lane&15)
#define WG_EPILOGUE(ACC, T)                                                    \
  {                                                                            \
    const int  col = Nsup + ((T) << 4) + l16;                                  \
    const bool cin = (col < N);                                                \
    const float bv = cin ? bias[col] : 0.f;                                    \
    if (Cmax) {                                                                \
      float m = -3.4e38f;                                                      \
      _Pragma("unroll")                                                        \
      for (int r = 0; r < 8; ++r) {                                            \
        float v = ACC[r] + bv;                                                 \
        if (relu) v = fmaxf(v, 0.f);                                           \
        m = fmaxf(m, v);                                                       \
      }                                                                        \
      m = fmaxf(m, __shfl_xor(m, 16, 32)); /* rows M+0..7 with M+8..15 */      \
      if (lane < 16 && cin)                                                    \
        atomicMax(&Cmax[(size_t)(Mbase >> gshift) * N + col], enc_ord(m));     \
    } else if (cin) {                                                          \
      _Pragma("unroll")                                                        \
      for (int r = 0; r < 8; ++r) {                                            \
        float v = ACC[r] + bv;                                                 \
        if (relu) v = fmaxf(v, 0.f);                                           \
        size_t idx = (size_t)(Mbase + rOff + r) * ldc + col;                   \
        if (half_out) ((_Float16*)Cout)[idx] = (_Float16)v;                    \
        else          ((float*)Cout)[idx]    = v;                              \
      }                                                                        \
    }                                                                          \
  }

  WG_EPILOGUE(acc0, 0)
  WG_EPILOGUE(acc1, 1)
  WG_EPILOGUE(acc2, 2)
  WG_EPILOGUE(acc3, 3)
#undef WG_EPILOGUE
}

// ---- small elementwise kernels --------------------------------------------
__global__ void decode_h_kernel(const unsigned* __restrict__ e, _Float16* __restrict__ out, int n) {
  int i = blockIdx.x * blockDim.x + threadIdx.x;
  if (i < n) out[i] = (_Float16)dec_ord(e[i]);
}

__global__ void decode_strided_h_kernel(const unsigned* __restrict__ e, _Float16* __restrict__ out,
                                        int N, int ld, int off, int total) {
  int i = blockIdx.x * blockDim.x + threadIdx.x;
  if (i < total) {
    int b = i / N, n = i - b * N;
    out[(size_t)b * ld + off + n] = (_Float16)dec_ord(e[i]);
  }
}

// z = mu_z + tril_softplus(res[5:],5) @ eps ; zc = [z, obj]   (f32 in/out)
__global__ void latent_kernel(const float* __restrict__ res, const float* __restrict__ eps,
                              const float* __restrict__ obj, float* __restrict__ zc) {
  int b = blockIdx.x * blockDim.x + threadIdx.x;
  if (b >= 64) return;
  const float* r = res + b * 20;
  float z[5];
  int idx = 0;
  for (int i = 0; i < 5; ++i) {
    float s = r[i];
    for (int j = 0; j <= i; ++j, ++idx) {
      float lij = r[5 + idx];                 // vec index i*(i+1)/2 + j
      if (j == i) lij = softplus_f(lij);
      s += lij * eps[b * 5 + j];
    }
    z[i] = s;
  }
  for (int i = 0; i < 5; ++i) zc[b * 10 + i]     = z[i];
  for (int i = 0; i < 5; ++i) zc[b * 10 + 5 + i] = obj[b * 5 + i];
}

// feat[b,s,:] = [push(5), zc(10), 0 pad to 32] as f16
__global__ void build_feat_kernel(const float* __restrict__ push, const float* __restrict__ zc,
                                  _Float16* __restrict__ feat) {
  int i = blockIdx.x * blockDim.x + threadIdx.x;
  if (i >= 64 * 512) return;
  int b = i >> 9;
  _Float16* f = feat + (size_t)i * 32;
  const float* p = push + (size_t)i * 5;
#pragma unroll
  for (int j = 0; j < 5; ++j)  f[j]     = (_Float16)p[j];
#pragma unroll
  for (int j = 0; j < 10; ++j) f[5 + j] = (_Float16)zc[b * 10 + j];
#pragma unroll
  for (int j = 15; j < 32; ++j) f[j] = (_Float16)0.f;
}

// out[b,s,:] = [mu(3), tril_softplus(v[3:],3) row-major (9)]
__global__ void final_kernel(const float* __restrict__ v9, float* __restrict__ out) {
  int i = blockIdx.x * blockDim.x + threadIdx.x;
  if (i >= 64 * 512) return;
  const float* v = v9 + (size_t)i * 9;
  float* o = out + (size_t)i * 12;
  o[0] = v[0]; o[1] = v[1]; o[2] = v[2];
  o[3] = softplus_f(v[3]); o[4]  = 0.f;              o[5]  = 0.f;
  o[6] = v[4];             o[7]  = softplus_f(v[5]); o[8]  = 0.f;
  o[9] = v[6];             o[10] = v[7];             o[11] = softplus_f(v[8]);
}

// ---- host ------------------------------------------------------------------
static inline int cdiv(int a, int b) { return (a + b - 1) / b; }

static void launch_gemm(hipStream_t s, const _Float16* A, const _Float16* Bpk, const float* bias,
                        void* Cout, unsigned* Cmax, int R, int K, int N, int ldc,
                        int relu, int gshift, int half_out) {
  int RT = R >> 4;
  int CT = (N + 63) >> 6;
  int blocks = cdiv(RT * CT, 8);           // 256 threads = 8 wave32 per block
  hipLaunchKernelGGL(wmma_gemm_kernel, dim3(blocks), dim3(256), 0, s,
                     A, Bpk, bias, Cout, Cmax, R, K, N, ldc, relu, gshift, half_out);
}

// pack one weight matrix; returns packed pointer, advances bump allocator
static _Float16* pack_w(hipStream_t s, const float* W, int K, int N, char*& bump) {
  int KC = cdiv(K, 32);
  int NT = cdiv(N, 64) * 4;                // tiles, padded to full 64-wide supertiles
  _Float16* P = (_Float16*)bump;
  size_t bytes = (size_t)NT * KC * 512 * sizeof(_Float16);
  bump += (bytes + 255) & ~(size_t)255;
  int total = NT * KC * 32;
  hipLaunchKernelGGL(pack_w_kernel, dim3(cdiv(total, 256)), dim3(256), 0, s, W, P, K, N, KC, NT);
  return P;
}

extern "C" void kernel_launch(void* const* d_in, const int* in_sizes, int n_in,
                              void* d_out, int out_size, void* d_ws, size_t ws_size,
                              hipStream_t stream) {
  (void)in_sizes; (void)n_in; (void)out_size; (void)ws_size;
  const float* mesh  = (const float*)d_in[0];
  const float* obj   = (const float*)d_in[1];
  const float* push  = (const float*)d_in[2];
  const float* npush = (const float*)d_in[3];
  const float* eps   = (const float*)d_in[4];
  const float* pn_w1 = (const float*)d_in[5];  const float* pn_b1 = (const float*)d_in[6];
  const float* pn_w2 = (const float*)d_in[7];  const float* pn_b2 = (const float*)d_in[8];
  const float* pn_w3 = (const float*)d_in[9];  const float* pn_b3 = (const float*)d_in[10];
  const float* fc_w  = (const float*)d_in[11]; const float* fc_b  = (const float*)d_in[12];
  const float* cw1   = (const float*)d_in[13]; const float* cb1   = (const float*)d_in[14];
  const float* cw2   = (const float*)d_in[15]; const float* cb2   = (const float*)d_in[16];
  const float* cw3   = (const float*)d_in[17]; const float* cb3   = (const float*)d_in[18];
  const float* l1_w  = (const float*)d_in[19]; const float* l1_b  = (const float*)d_in[20];
  const float* l2_w  = (const float*)d_in[21]; const float* l2_b  = (const float*)d_in[22];
  const float* l3_w  = (const float*)d_in[23]; const float* l3_b  = (const float*)d_in[24];
  const float* d1_w  = (const float*)d_in[25]; const float* d1_b  = (const float*)d_in[26];
  const float* d2_w  = (const float*)d_in[27]; const float* d2_b  = (const float*)d_in[28];
  const float* d3_w  = (const float*)d_in[29]; const float* d3_b  = (const float*)d_in[30];

  // deterministic bump allocator over d_ws (256B aligned blocks)
  char* bump = (char*)d_ws;
  auto alloc = [&](size_t bytes) -> void* {
    void* r = bump;
    bump += (bytes + 255) & ~(size_t)255;
    return r;
  };

  // f16 activation buffers (K padded to %32)
  _Float16* MESHP  = (_Float16*)alloc(65536ull * 32 * 2);
  _Float16* NPUSHP = (_Float16*)alloc(32768ull * 32 * 2);
  _Float16* FEATP  = (_Float16*)alloc(32768ull * 32 * 2);
  _Float16* PN1    = (_Float16*)alloc(65536ull * 64 * 2);
  _Float16* PN2    = (_Float16*)alloc(65536ull * 128 * 2);
  _Float16* MMAXH  = (_Float16*)alloc(64ull * 1024 * 2);
  _Float16* H      = (_Float16*)alloc(64ull * 1536 * 2);
  _Float16* C1     = (_Float16*)alloc(32768ull * 64 * 2);
  _Float16* C2     = (_Float16*)alloc(32768ull * 256 * 2);
  _Float16* H2     = (_Float16*)alloc(64ull * 256 * 2);
  _Float16* H3     = (_Float16*)alloc(64ull * 128 * 2);
  _Float16* D1     = (_Float16*)alloc(32768ull * 256 * 2);
  _Float16* D2     = (_Float16*)alloc(32768ull * 1024 * 2);
  unsigned* MMAXE  = (unsigned*)alloc(64ull * 1024 * 4);
  unsigned* PMAXE  = (unsigned*)alloc(64ull * 1024 * 4);
  float*    RES    = (float*)alloc(64ull * 20 * 4);
  float*    ZC     = (float*)alloc(64ull * 10 * 4);
  float*    D3     = (float*)alloc(32768ull * 9 * 4);

  // packed weights (f16 B-fragment layout), ~3.4 MB total
  _Float16* Ppn1 = pack_w(stream, pn_w1,    6,   64, bump);
  _Float16* Ppn2 = pack_w(stream, pn_w2,   64,  128, bump);
  _Float16* Ppn3 = pack_w(stream, pn_w3,  128, 1024, bump);
  _Float16* Pfc  = pack_w(stream, fc_w,  1024,  512, bump);
  _Float16* Pc1  = pack_w(stream, cw1,      5,   64, bump);
  _Float16* Pc2  = pack_w(stream, cw2,     64,  256, bump);
  _Float16* Pc3  = pack_w(stream, cw3,    256, 1024, bump);
  _Float16* Pl1  = pack_w(stream, l1_w,  1536,  256, bump);
  _Float16* Pl2  = pack_w(stream, l2_w,   256,  128, bump);
  _Float16* Pl3  = pack_w(stream, l3_w,   128,   20, bump);
  _Float16* Pd1  = pack_w(stream, d1_w,    15,  256, bump);
  _Float16* Pd2  = pack_w(stream, d2_w,   256, 1024, bump);
  _Float16* Pd3  = pack_w(stream, d3_w,  1024,    9, bump);

  // pad/convert raw inputs to f16
  pad_input_kernel<<<cdiv(65536, 256), 256, 0, stream>>>(mesh,  MESHP,  65536, 6, 32);
  pad_input_kernel<<<cdiv(32768, 256), 256, 0, stream>>>(npush, NPUSHP, 32768, 5, 32);

  // init encoded-max buffers (key 0 is below every real float's key)
  hipMemsetAsync(MMAXE, 0, 64 * 1024 * sizeof(unsigned), stream);
  hipMemsetAsync(PMAXE, 0, 64 * 1024 * sizeof(unsigned), stream);

  // --- PointNet branch: 65536 rows; relu+max fused into pn3 epilogue ---
  launch_gemm(stream, MESHP, Ppn1, pn_b1, PN1, nullptr, 65536,  32,   64,   64, 1, 0, 1);
  launch_gemm(stream, PN1,   Ppn2, pn_b2, PN2, nullptr, 65536,  64,  128,  128, 1, 0, 1);
  launch_gemm(stream, PN2,   Ppn3, pn_b3, nullptr, MMAXE, 65536, 128, 1024, 0, 1, 10, 0);
  decode_h_kernel<<<cdiv(65536, 256), 256, 0, stream>>>(MMAXE, MMAXH, 65536);
  launch_gemm(stream, MMAXH, Pfc, fc_b, H, nullptr, 64, 1024, 512, 1536, 0, 0, 1); // H[:, :512]

  // --- Push encoder branch: 32768 rows; max (no relu) fused into c3 ---
  launch_gemm(stream, NPUSHP, Pc1, cb1, C1, nullptr, 32768,  32,   64,  64, 1, 0, 1);
  launch_gemm(stream, C1,     Pc2, cb2, C2, nullptr, 32768,  64,  256, 256, 1, 0, 1);
  launch_gemm(stream, C2,     Pc3, cb3, nullptr, PMAXE, 32768, 256, 1024, 0, 0, 9, 0);
  decode_strided_h_kernel<<<cdiv(65536, 256), 256, 0, stream>>>(PMAXE, H, 1024, 1536, 512, 65536);

  // --- Head MLP (tiny) + latent sample ---
  launch_gemm(stream, H,  Pl1, l1_b, H2,  nullptr, 64, 1536, 256, 256, 1, 0, 1);
  launch_gemm(stream, H2, Pl2, l2_b, H3,  nullptr, 64,  256, 128, 128, 1, 0, 1);
  launch_gemm(stream, H3, Pl3, l3_b, RES, nullptr, 64,  128,  20,  20, 0, 0, 0); // f32 out
  latent_kernel<<<1, 64, 0, stream>>>(RES, eps, obj, ZC);

  // --- Decoder: 32768 rows ---
  build_feat_kernel<<<cdiv(32768, 256), 256, 0, stream>>>(push, ZC, FEATP);
  launch_gemm(stream, FEATP, Pd1, d1_b, D1, nullptr, 32768,   32,  256,  256, 1, 0, 1);
  launch_gemm(stream, D1,    Pd2, d2_b, D2, nullptr, 32768,  256, 1024, 1024, 1, 0, 1);
  launch_gemm(stream, D2,    Pd3, d3_b, D3, nullptr, 32768, 1024,    9,    9, 0, 0, 0); // f32 out
  final_kernel<<<cdiv(32768, 256), 256, 0, stream>>>(D3, (float*)d_out);
}
